// puremamba_model_78340203479324
// MI455X (gfx1250) — matrix operand order repaired and checked by reading, not compiled, not run on hardware
//
#include <hip/hip_runtime.h>
#include <hip/hip_bf16.h>

// ---------------- problem constants ----------------
constexpr int Bn  = 8;
constexpr int Ln  = 2048;
constexpr int Dn  = 128;
constexpr int NLn = 4;
constexpr int DSn = 16;
constexpr int DCn = 3;
constexpr int DIn = 256;       // EXP*D
constexpr int DRn = 8;         // (D+15)/16
constexpr int NCn = 10;
constexpr int Mrows = Bn * Ln; // 16384 token rows
constexpr float EPSc = 1e-5f;

typedef __attribute__((ext_vector_type(16))) _Float16 v16h;
typedef __attribute__((ext_vector_type(8)))  float    v8f;

union V16H { v16h v; uint4 q[2]; };

__device__ __forceinline__ float silu_f(float x) {
  return x * (1.0f / (1.0f + __expf(-x)));
}
__device__ __forceinline__ float softplus_f(float x) {
  return (x > 20.0f) ? x : log1pf(__expf(x));
}

// ---------------- f32 -> f16 weight conversion ----------------
__global__ void f32_to_f16_kernel(const float* __restrict__ src,
                                  _Float16* __restrict__ dst, int n) {
  int i = blockIdx.x * blockDim.x + threadIdx.x;
  if (i < n) dst[i] = (_Float16)src[i];
}

// ---------------- h = transpose(x, (0,2,1)) : (B,D,L) -> (B,L,D) ----------------
__global__ void transpose_in_kernel(const float* __restrict__ x,
                                    float* __restrict__ h) {
  int idx = blockIdx.x * blockDim.x + threadIdx.x;
  if (idx >= Bn * Ln * Dn) return;
  int dd = idx % Dn;
  int l  = (idx / Dn) % Ln;
  int b  = idx / (Dn * Ln);
  h[idx] = x[((size_t)b * Dn + dd) * Ln + l];
}

// ---------------- rmsnorm over D=128, write f16 ----------------
__global__ void rmsnorm_kernel(const float* __restrict__ h,
                               const float* __restrict__ w,
                               _Float16* __restrict__ xn) {
  __shared__ float red[Dn];
  int row = blockIdx.x;
  int d   = threadIdx.x;
  float v = h[(size_t)row * Dn + d];
  red[d] = v * v;
  __syncthreads();
  for (int s = Dn / 2; s > 0; s >>= 1) {
    if (d < s) red[d] += red[d + s];
    __syncthreads();
  }
  float rs = rsqrtf(red[0] * (1.0f / Dn) + EPSc);
  xn[(size_t)row * Dn + d] = (_Float16)(v * rs * w[d]);
}

// ---------------- WMMA GEMM: C[M,N] (+)= A[M,K] * W[N,K]^T ----------------
// A: f16 row-major (K contiguous), W: f16 row-major N x K, C: f32 row-major.
// Register tiling: one wave -> 64(M) x 16(N) outputs = 4 WMMAs per K-step,
// reusing the single B (weight) fragment across 4 A fragments.
// blockDim = 128 (4 waves along N). Requires M % 64 == 0, N % 64 == 0.
template <bool RESIDUAL>
__global__ void gemm_wmma_f16(const _Float16* __restrict__ A,
                              const _Float16* __restrict__ W,
                              float* __restrict__ C,
                              int N, int K) {
  const int lane  = threadIdx.x & 31;
  const int wave  = threadIdx.x >> 5;
  const int tileM = blockIdx.x;               // 64 rows
  const int tileN = blockIdx.y * 4 + wave;    // 16 cols per wave
  const int l15   = lane & 15;
  const int hi    = lane >> 4;                // lane half
  const int kbA   = hi * 8;                   // A frag: K 0..7 / 8..15 (+16 chunk)
  const int kbB   = hi * 8;                   // B frag: K 0..7 / 8..15 (+16 chunk)

  const _Float16* ap0 = A + (size_t)(tileM * 64 +  0 + l15) * K + kbA;
  const _Float16* ap1 = A + (size_t)(tileM * 64 + 16 + l15) * K + kbA;
  const _Float16* ap2 = A + (size_t)(tileM * 64 + 32 + l15) * K + kbA;
  const _Float16* ap3 = A + (size_t)(tileM * 64 + 48 + l15) * K + kbA;
  const _Float16* bp  = W + (size_t)(tileN * 16 + l15) * K + kbB;

  v8f acc0 = {}, acc1 = {}, acc2 = {}, acc3 = {};
  for (int k0 = 0; k0 < K; k0 += 32) {
    V16H b;
    b.q[0] = *(const uint4*)(bp + k0);        // K = k0+kbB    .. +7
    b.q[1] = *(const uint4*)(bp + k0 + 16);   // K = k0+kbB+16 .. +23
    V16H a0, a1, a2, a3;
    a0.q[0] = *(const uint4*)(ap0 + k0);
    a0.q[1] = *(const uint4*)(ap0 + k0 + 16);
    a1.q[0] = *(const uint4*)(ap1 + k0);
    a1.q[1] = *(const uint4*)(ap1 + k0 + 16);
    a2.q[0] = *(const uint4*)(ap2 + k0);
    a2.q[1] = *(const uint4*)(ap2 + k0 + 16);
    a3.q[0] = *(const uint4*)(ap3 + k0);
    a3.q[1] = *(const uint4*)(ap3 + k0 + 16);
    __builtin_prefetch(ap0 + k0 + 32, 0, 1);
    __builtin_prefetch(ap2 + k0 + 32, 0, 1);
    acc0 = __builtin_amdgcn_wmma_f32_16x16x32_f16(false, a0.v, false, b.v,
                                                  (short)0, acc0, false, false);
    acc1 = __builtin_amdgcn_wmma_f32_16x16x32_f16(false, a1.v, false, b.v,
                                                  (short)0, acc1, false, false);
    acc2 = __builtin_amdgcn_wmma_f32_16x16x32_f16(false, a2.v, false, b.v,
                                                  (short)0, acc2, false, false);
    acc3 = __builtin_amdgcn_wmma_f32_16x16x32_f16(false, a3.v, false, b.v,
                                                  (short)0, acc3, false, false);
  }

  // C/D layout: VGPR v holds M = v + 8*hi, N = lane&15
  const int col = tileN * 16 + l15;
  v8f accs[4] = {acc0, acc1, acc2, acc3};
#pragma unroll
  for (int i = 0; i < 4; ++i) {
    const int row0 = tileM * 64 + i * 16 + hi * 8;
#pragma unroll
    for (int v = 0; v < 8; ++v) {
      size_t idx = (size_t)(row0 + v) * N + col;
      if (RESIDUAL) C[idx] += accs[i][v];
      else          C[idx]  = accs[i][v];
    }
  }
}

// ---------------- causal depthwise conv (DC=3) + bias + SiLU ----------------
__global__ void conv_silu_kernel(const float* __restrict__ xz, // (B,L,2*DI)
                                 const float* __restrict__ cw, // (DI,3)
                                 const float* __restrict__ cb, // (DI)
                                 float* __restrict__ xc) {     // (B,L,DI)
  int idx = blockIdx.x * blockDim.x + threadIdx.x;
  if (idx >= Bn * Ln * DIn) return;
  int c = idx % DIn;
  int t = (idx / DIn) % Ln;
  int b = idx / (DIn * Ln);
  float w0 = cw[c * 3 + 0], w1 = cw[c * 3 + 1], w2 = cw[c * 3 + 2];
  size_t rb = ((size_t)b * Ln + t) * (2 * DIn) + c;
  float s = cb[c] + w2 * xz[rb];
  if (t >= 1) s += w1 * xz[rb - 2 * DIn];
  if (t >= 2) s += w0 * xz[rb - 4 * DIn];
  xc[idx] = silu_f(s);
}

// ---------------- x_proj: dbc[row,0..39] = xc_row . xp_w[o,:] ----------------
__global__ void xproj_kernel(const float* __restrict__ xc,   // (B*L, DI)
                             const float* __restrict__ xpw,  // (40, DI)
                             float* __restrict__ dbc) {      // (B*L, 40)
  __shared__ float srow[DIn];
  int row = blockIdx.x;
  for (int k = threadIdx.x; k < DIn; k += blockDim.x)
    srow[k] = xc[(size_t)row * DIn + k];
  __syncthreads();
  int o = threadIdx.x;
  if (o < DRn + 2 * DSn) {
    const float* wrow = xpw + (size_t)o * DIn;
    float s = 0.f;
    for (int k = 0; k < DIn; ++k) s += srow[k] * wrow[k];
    dbc[(size_t)row * 40 + o] = s;
  }
}

// ---------------- delta = softplus(dbc[:, :8] @ dt_w^T + dt_b) ----------------
__global__ void delta_kernel(const float* __restrict__ dbc,  // (B*L,40)
                             const float* __restrict__ dtw,  // (DI,8)
                             const float* __restrict__ dtb,  // (DI)
                             float* __restrict__ delta) {    // (B*L,DI)
  int idx = blockIdx.x * blockDim.x + threadIdx.x;
  if (idx >= Mrows * DIn) return;
  int d   = idx % DIn;
  int row = idx / DIn;
  const float* dr = dbc + (size_t)row * 40;
  const float* wr = dtw + (size_t)d * DRn;
  float s = dtb[d];
#pragma unroll
  for (int r = 0; r < DRn; ++r) s += dr[r] * wr[r];
  delta[idx] = softplus_f(s);
}

// ---------------- selective scan: per (b, channel), sequential in L ----------------
__global__ void scan_kernel(const float* __restrict__ xc,     // (B,L,DI)
                            const float* __restrict__ delta,  // (B,L,DI)
                            const float* __restrict__ dbc,    // (B,L,40) B@8..23 C@24..39
                            const float* __restrict__ A_log,  // (DI,DS)
                            const float* __restrict__ Dp,     // (DI)
                            float* __restrict__ y) {          // (B,L,DI)
  const int b = blockIdx.x;
  const int d = threadIdx.x; // 256 channels
  __shared__ float sB[64][DSn];
  __shared__ float sC[64][DSn];
  float A[DSn];
#pragma unroll
  for (int s = 0; s < DSn; ++s) A[s] = -__expf(A_log[d * DSn + s]);
  const float Dv = Dp[d];
  float hst[DSn];
#pragma unroll
  for (int s = 0; s < DSn; ++s) hst[s] = 0.f;

  for (int t0 = 0; t0 < Ln; t0 += 64) {
    __syncthreads();
    for (int e = threadIdx.x; e < 64 * DSn; e += DIn) {
      int tt = e >> 4, s = e & 15;
      size_t rb = ((size_t)b * Ln + t0 + tt) * 40;
      sB[tt][s] = dbc[rb + 8 + s];
      sC[tt][s] = dbc[rb + 24 + s];
    }
    __syncthreads();
    for (int tt = 0; tt < 64; ++tt) {
      size_t idx = ((size_t)b * Ln + t0 + tt) * DIn + d;
      float dl = delta[idx];
      float u  = xc[idx];
      float du = dl * u;
      float acc = 0.f;
#pragma unroll
      for (int s = 0; s < DSn; ++s) {
        hst[s] = __expf(dl * A[s]) * hst[s] + du * sB[tt][s];
        acc += hst[s] * sC[tt][s];
      }
      y[idx] = acc + u * Dv;
    }
  }
}

// ---------------- yz16 = f16( y * silu(z) ) ----------------
__global__ void gate_kernel(const float* __restrict__ y,   // (B*L,DI)
                            const float* __restrict__ xz,  // (B*L,2*DI)
                            _Float16* __restrict__ yz16) {
  int idx = blockIdx.x * blockDim.x + threadIdx.x;
  if (idx >= Mrows * DIn) return;
  int c   = idx % DIn;
  int row = idx / DIn;
  float z = xz[(size_t)row * (2 * DIn) + DIn + c];
  yz16[idx] = (_Float16)(y[idx] * silu_f(z));
}

// ---------------- pooled[b,l] = mean_d h[b,l,d] ----------------
__global__ void pooled_kernel(const float* __restrict__ h,
                              float* __restrict__ pooled) {
  int idx = blockIdx.x * blockDim.x + threadIdx.x;
  if (idx >= Bn * Ln) return;
  const float* r = h + (size_t)idx * Dn;
  float s = 0.f;
  for (int d = 0; d < Dn; ++d) s += r[d];
  pooled[idx] = s * (1.0f / Dn);
}

// ---------------- out[b,n] = pooled[b,:] . fc_w[n,:] + fc_b[n] ----------------
__global__ void fc_kernel(const float* __restrict__ pooled,
                          const float* __restrict__ fcw,
                          const float* __restrict__ fcb,
                          float* __restrict__ out) {
  __shared__ float red[256];
  int b = blockIdx.x / NCn;
  int n = blockIdx.x % NCn;
  float s = 0.f;
  for (int l = threadIdx.x; l < Ln; l += 256)
    s += pooled[(size_t)b * Ln + l] * fcw[(size_t)n * Ln + l];
  red[threadIdx.x] = s;
  __syncthreads();
  for (int st = 128; st > 0; st >>= 1) {
    if (threadIdx.x < st) red[threadIdx.x] += red[threadIdx.x + st];
    __syncthreads();
  }
  if (threadIdx.x == 0) out[b * NCn + n] = red[0] + fcb[n];
}

// ---------------- host launch ----------------
extern "C" void kernel_launch(void* const* d_in, const int* in_sizes, int n_in,
                              void* d_out, int out_size, void* d_ws, size_t ws_size,
                              hipStream_t stream) {
  const float* x       = (const float*)d_in[0];
  const float* norm_w  = (const float*)d_in[1];
  const float* in_w    = (const float*)d_in[2];
  const float* conv_w  = (const float*)d_in[3];
  const float* conv_b  = (const float*)d_in[4];
  const float* xp_w    = (const float*)d_in[5];
  const float* dt_w    = (const float*)d_in[6];
  const float* dt_b    = (const float*)d_in[7];
  const float* A_log   = (const float*)d_in[8];
  const float* D_param = (const float*)d_in[9];
  const float* out_w   = (const float*)d_in[10];
  const float* fc_w    = (const float*)d_in[11];
  const float* fc_b    = (const float*)d_in[12];
  float* out = (float*)d_out;

  char* wsp = (char*)d_ws;
  auto carve = [&](size_t bytes) -> char* {
    char* p = wsp;
    wsp += (bytes + 255) & ~(size_t)255;
    return p;
  };
  float*    h      = (float*)   carve((size_t)Mrows * Dn  * 4);
  _Float16* xn16   = (_Float16*)carve((size_t)Mrows * Dn  * 2);
  float*    xz     = (float*)   carve((size_t)Mrows * 2 * DIn * 4);
  float*    xc     = (float*)   carve((size_t)Mrows * DIn * 4);
  float*    dbc    = (float*)   carve((size_t)Mrows * 40  * 4);
  float*    deltab = (float*)   carve((size_t)Mrows * DIn * 4);
  float*    yv     = (float*)   carve((size_t)Mrows * DIn * 4);
  _Float16* yz16   = (_Float16*)carve((size_t)Mrows * DIn * 2);
  float*    pooled = (float*)   carve((size_t)Bn * Ln * 4);
  _Float16* inW16  = (_Float16*)carve((size_t)NLn * 2 * DIn * Dn * 2);
  _Float16* outW16 = (_Float16*)carve((size_t)NLn * Dn * DIn * 2);

  // weights -> f16 (every call; no caching allowed)
  {
    int nIn = NLn * 2 * DIn * Dn;
    f32_to_f16_kernel<<<(nIn + 255) / 256, 256, 0, stream>>>(in_w, inW16, nIn);
    int nOut = NLn * Dn * DIn;
    f32_to_f16_kernel<<<(nOut + 255) / 256, 256, 0, stream>>>(out_w, outW16, nOut);
  }

  // h = transpose(x)
  transpose_in_kernel<<<(Mrows * Dn + 255) / 256, 256, 0, stream>>>(x, h);

  int ne = Mrows * DIn;
  for (int l = 0; l < NLn; ++l) {
    rmsnorm_kernel<<<Mrows, Dn, 0, stream>>>(h, norm_w + l * Dn, xn16);

    // in_proj: (16384 x 128) @ (512 x 128)^T -> xz
    gemm_wmma_f16<false><<<dim3(Mrows / 64, (2 * DIn) / 64), 128, 0, stream>>>(
        xn16, inW16 + (size_t)l * 2 * DIn * Dn, xz, 2 * DIn, Dn);

    conv_silu_kernel<<<(ne + 255) / 256, 256, 0, stream>>>(
        xz, conv_w + (size_t)l * DIn * DCn, conv_b + (size_t)l * DIn, xc);

    xproj_kernel<<<Mrows, 64, 0, stream>>>(
        xc, xp_w + (size_t)l * (DRn + 2 * DSn) * DIn, dbc);

    delta_kernel<<<(ne + 255) / 256, 256, 0, stream>>>(
        dbc, dt_w + (size_t)l * DIn * DRn, dt_b + (size_t)l * DIn, deltab);

    scan_kernel<<<Bn, DIn, 0, stream>>>(
        xc, deltab, dbc, A_log + (size_t)l * DIn * DSn, D_param + (size_t)l * DIn, yv);

    gate_kernel<<<(ne + 255) / 256, 256, 0, stream>>>(yv, xz, yz16);

    // out_proj: (16384 x 256) @ (128 x 256)^T -> += h (residual)
    gemm_wmma_f16<true><<<dim3(Mrows / 64, Dn / 64), 128, 0, stream>>>(
        yz16, outW16 + (size_t)l * Dn * DIn, h, Dn, DIn);
  }

  pooled_kernel<<<(Bn * Ln + 255) / 256, 256, 0, stream>>>(h, pooled);
  fc_kernel<<<Bn * NCn, 256, 0, stream>>>(pooled, fc_w, fc_b, out);
}
